// DREAM_50251117363953
// MI455X (gfx1250) — compile-verified, matrix operand order checked by reading, not run on hardware
//
#include <hip/hip_runtime.h>
#include <hip/hip_bf16.h>
#include <math.h>

typedef __attribute__((ext_vector_type(16))) __bf16 v16bf;
typedef __attribute__((ext_vector_type(8)))  __bf16 v8bf;
typedef __attribute__((ext_vector_type(8)))  float  v8f;
typedef __attribute__((address_space(3)))    char   lds_char;

static constexpr int Bb = 64, Tt = 512, Ii = 1024, Hh = 2048, Rr = 8;
static constexpr int Mm = Bb * Tt;   // 32768 tokens

__device__ __forceinline__ v16bf cat8(v8bf lo, v8bf hi) {
  return __builtin_shufflevector(lo, hi, 0,1,2,3,4,5,6,7,8,9,10,11,12,13,14,15);
}

// ---- CDNA5 async global->LDS staging (ASYNCcnt path, bypasses VGPRs) ----
__device__ __forceinline__ void async_copy_b128(void* lds_dst, const void* gsrc) {
  unsigned loff = (unsigned)(unsigned long long)(lds_char*)lds_dst;
  asm volatile("global_load_async_to_lds_b128 %0, %1, off"
               :: "v"(loff), "v"(gsrc) : "memory");
}
__device__ __forceinline__ void wait_async0() {
  asm volatile("s_wait_asynccnt 0x0" ::: "memory");
}

// ---------------------------------------------------------------- conversions
__global__ __launch_bounds__(256) void f32_to_bf16(const float* __restrict__ in,
                                                   __bf16* __restrict__ out, size_t n4) {
  size_t i = (size_t)blockIdx.x * blockDim.x + threadIdx.x;
  size_t stride = (size_t)gridDim.x * blockDim.x;
  for (; i < n4; i += stride) {
    float4 f = ((const float4*)in)[i];
    __bf16* o = out + i * 4;
    o[0] = (__bf16)f.x; o[1] = (__bf16)f.y; o[2] = (__bf16)f.z; o[3] = (__bf16)f.w;
  }
}

// W (K x N, f32) -> Wt (N x K, bf16)
__global__ __launch_bounds__(256) void transpose_bf16(const float* __restrict__ W,
                                                      __bf16* __restrict__ Wt,
                                                      int K, int N) {
  __shared__ float tile[32][33];
  const int kb = blockIdx.x * 32, nb = blockIdx.y * 32;
  const int tx = threadIdx.x & 31, ty = threadIdx.x >> 5;
  for (int i = ty; i < 32; i += 8)
    tile[i][tx] = W[(size_t)(kb + i) * N + nb + tx];
  __syncthreads();
  for (int i = ty; i < 32; i += 8)
    Wt[(size_t)(nb + i) * K + kb + tx] = (__bf16)tile[tx][i];
}

// ------------------------------------------------- WMMA GEMM: C = A * Bt^T (+bias)
// A: M x K (bf16 row-major), Bt: N x K (bf16 row-major), C: M x N (f32)
__global__ __launch_bounds__(256) void gemm_bf16(const __bf16* __restrict__ A,
                                                 const __bf16* __restrict__ Bt,
                                                 const float* __restrict__ bias,
                                                 float* __restrict__ C,
                                                 int Ndim, int Kdim) {
  __shared__ __bf16 As[64 * 64];
  __shared__ __bf16 Bs[32 * 64];
  const int tid  = threadIdx.x;
  const int lane = tid & 31, wave = tid >> 5;
  const int mi = wave >> 1, ni = wave & 1;              // 4 x 2 wave grid -> 64x32 tile
  const int mb = blockIdx.y * 64, nb = blockIdx.x * 32;
  const int arow = mi * 16 + (lane & 15);
  const int brow = ni * 16 + (lane & 15);
  const int koff = (lane >> 4) << 3;                    // 0 or 8 (ISA 16-bit A/B layout)

  v8f acc = {};
  for (int kb = 0; kb < Kdim; kb += 64) {
    __syncthreads();
#pragma unroll
    for (int p = 0; p < 2; ++p) {                       // A chunk: 64x64 bf16, async->LDS
      int i = tid + p * 256;
      int r = i >> 3, q = i & 7;
      async_copy_b128(As + r * 64 + q * 8, A + (size_t)(mb + r) * Kdim + kb + q * 8);
    }
    {                                                   // B chunk: 32x64 bf16, async->LDS
      int r = tid >> 3, q = tid & 7;
      async_copy_b128(Bs + r * 64 + q * 8, Bt + (size_t)(nb + r) * Kdim + kb + q * 8);
    }
    if (kb + 64 < Kdim)                                 // global_prefetch_b8 of next chunk
      __builtin_prefetch(A + (size_t)(mb + (tid >> 2)) * Kdim + kb + 64, 0, 1);
    wait_async0();
    __syncthreads();
#pragma unroll
    for (int ks = 0; ks < 2; ++ks) {
      const int k0 = ks * 32;
      v8bf alo = *(const v8bf*)(As + arow * 64 + k0 + koff);
      v8bf ahi = *(const v8bf*)(As + arow * 64 + k0 + koff + 16);
      v8bf blo = *(const v8bf*)(Bs + brow * 64 + k0 + koff);
      v8bf bhi = *(const v8bf*)(Bs + brow * 64 + k0 + koff + 16);
      acc = __builtin_amdgcn_wmma_f32_16x16x32_bf16(
          false, cat8(alo, ahi), false, cat8(blo, bhi), (short)0, acc, false, false);
    }
  }
  const int hi2 = lane >> 4, nc = lane & 15;
  const int col = nb + ni * 16 + nc;
  const float bia = bias ? bias[col] : 0.0f;
#pragma unroll
  for (int g = 0; g < 8; ++g) {                         // C layout: m = g + 8*hi, n = lane%16
    int rowo = mb + mi * 16 + 8 * hi2 + g;
    C[(size_t)rowo * Ndim + col] = acc[g] + bia;
  }
}

// ---------------------------------------------- rank-8 projections (k, lam raw)
__global__ __launch_bounds__(256) void proj_small(const float* __restrict__ x,
                                                  const float* __restrict__ Wk,
                                                  const float* __restrict__ Wl,
                                                  float* __restrict__ kraw,
                                                  float* __restrict__ lraw) {
  __shared__ float xs[16 * 256];
  const int tokBase = blockIdx.x * 16;
  const int tslot = threadIdx.x >> 4;       // token within block
  const int slot  = threadIdx.x & 15;       // 0..7 -> k, 8..15 -> lam
  const float* W = (slot < 8) ? Wk : Wl;
  const int r = slot & 7;
  float acc = 0.f;
  for (int c = 0; c < Ii; c += 256) {
    __syncthreads();
    for (int i = threadIdx.x; i < 16 * 64; i += 256) {  // 16 rows x 64 float4, async->LDS
      int row = i >> 6, q = i & 63;
      async_copy_b128(xs + row * 256 + q * 4,
                      x + (size_t)(tokBase + row) * Ii + c + q * 4);
    }
    wait_async0();
    __syncthreads();
    const float* xr = xs + tslot * 256;
    for (int i = 0; i < 256; ++i) acc += xr[i] * W[(size_t)(c + i) * Rr + r];
  }
  size_t tok = (size_t)tokBase + tslot;
  if (slot < 8) kraw[tok * Rr + r] = acc;
  else          lraw[tok * Rr + r] = acc;
}

__global__ __launch_bounds__(256) void finalize_klam(const float* __restrict__ kraw,
                                                     const float* __restrict__ lraw,
                                                     const float* __restrict__ b_lam,
                                                     float* __restrict__ kn,
                                                     float* __restrict__ ln) {
  int tok = blockIdx.x * 256 + threadIdx.x;
  if (tok >= Mm) return;
  float kv[Rr]; float s = 0.f;
#pragma unroll
  for (int r = 0; r < Rr; ++r) {
    float z = kraw[(size_t)tok * Rr + r];
    float e = (z > 0.f) ? (z + 1.f) : expf(z);          // elu(z)+1
    kv[r] = e; s += e;
  }
  float inv = 1.f / s;
#pragma unroll
  for (int r = 0; r < Rr; ++r) kn[(size_t)tok * Rr + r] = kv[r] * inv;
#pragma unroll
  for (int r = 0; r < Rr; ++r) {
    float z = lraw[(size_t)tok * Rr + r] + b_lam[r];
    ln[(size_t)tok * Rr + r] = 1.f / (1.f + expf(-z));
  }
}

// ------------------------------------------------- persistent recurrence kernel
// grid = H/32 = 64 WGs; each WG owns 32 columns of H. Per step:
//   h_new = tanh(xp_t + h @ W_h + F.k_t) ; F = F*lam_t + v_t (x) k_t
__global__ __launch_bounds__(256, 1) void dream_rec(const __bf16* __restrict__ Wht,
                                                    const float* __restrict__ xp,
                                                    const float* __restrict__ vp,
                                                    const float* __restrict__ kn,
                                                    const float* __restrict__ ln,
                                                    __bf16* hbuf0, __bf16* hbuf1,
                                                    float* __restrict__ F,
                                                    float* __restrict__ out,
                                                    unsigned* bar) {
  extern __shared__ char smem[];
  __bf16* sWh = (__bf16*)smem;                               // 32 x 2048 bf16 (128 KB)
  __bf16* sH  = (__bf16*)(smem + 32 * 2048 * 2);             // 64 x 512 bf16 (64 KB)
  float*  sK  = (float*)(smem + 32 * 2048 * 2 + 64 * 512 * 2);
  float*  sL  = sK + 64 * 8;

  const int tid  = threadIdx.x;
  const int lane = tid & 31, wave = tid >> 5;
  const int mi = wave >> 1, ni = wave & 1;
  const int n0 = blockIdx.x * 32;
  const int arow = mi * 16 + (lane & 15);
  const int brow = ni * 16 + (lane & 15);
  const int koff = (lane >> 4) << 3;
  const int hi2 = lane >> 4, nc = lane & 15;
  const int ncol = n0 + ni * 16 + nc;

  // Cache this WG's W_h^T slice in LDS once for all 512 steps (async TDM-style copy).
  for (int i = tid; i < 32 * 2048 / 8; i += 256) {
    int n = i >> 8, q = i & 255;
    async_copy_b128(sWh + n * 2048 + q * 8, Wht + (size_t)(n0 + n) * Hh + q * 8);
  }
  wait_async0();

  for (int t = 0; t < Tt; ++t) {
    const __bf16* hc = (t & 1) ? hbuf1 : hbuf0;
    __bf16*       hn = (t & 1) ? hbuf0 : hbuf1;

    if (tid < 64) {                                          // stage k_t, lam_t (64 x 8)
      size_t tok = (size_t)tid * Tt + t;
      ((float4*)(sK + tid * 8))[0] = ((const float4*)(kn + tok * 8))[0];
      ((float4*)(sK + tid * 8))[1] = ((const float4*)(kn + tok * 8))[1];
      ((float4*)(sL + tid * 8))[0] = ((const float4*)(ln + tok * 8))[0];
      ((float4*)(sL + tid * 8))[1] = ((const float4*)(ln + tok * 8))[1];
    }

    v8f acc = {};
    for (int kc = 0; kc < 4; ++kc) {                         // K = 2048 in 4 x 512 chunks
      __syncthreads();
      for (int i = tid; i < 4096; i += 256) {                // h chunk: 64 x 512 bf16
        int r = i >> 6, q = i & 63;
        async_copy_b128(sH + r * 512 + q * 8,
                        hc + (size_t)r * Hh + kc * 512 + q * 8);
      }
      wait_async0();
      __syncthreads();
      const __bf16* wb = sWh + brow * 2048 + kc * 512;
#pragma unroll 4
      for (int ks = 0; ks < 16; ++ks) {
        const int k0 = ks * 32;
        v8bf alo = *(const v8bf*)(sH + arow * 512 + k0 + koff);
        v8bf ahi = *(const v8bf*)(sH + arow * 512 + k0 + koff + 16);
        v8bf blo = *(const v8bf*)(wb + k0 + koff);
        v8bf bhi = *(const v8bf*)(wb + k0 + koff + 16);
        acc = __builtin_amdgcn_wmma_f32_16x16x32_bf16(
            false, cat8(alo, ahi), false, cat8(blo, bhi), (short)0, acc, false, false);
      }
    }

    // Epilogue: readout + tanh + output + h_next + F update (lane owns (b,ncol)).
#pragma unroll
    for (int g = 0; g < 8; ++g) {
      const int b = mi * 16 + 8 * hi2 + g;
      const size_t tok = (size_t)b * Tt + t;
      float* Fp = F + ((size_t)b * Hh + ncol) * 8;
      float4 f0 = ((float4*)Fp)[0], f1 = ((float4*)Fp)[1];
      float4 k0v = ((float4*)(sK + b * 8))[0], k1v = ((float4*)(sK + b * 8))[1];
      float fast = f0.x * k0v.x + f0.y * k0v.y + f0.z * k0v.z + f0.w * k0v.w
                 + f1.x * k1v.x + f1.y * k1v.y + f1.z * k1v.z + f1.w * k1v.w;
      float val = tanhf(xp[tok * Hh + ncol] + acc[g] + fast);
      out[tok * Hh + ncol] = val;
      hn[(size_t)b * Hh + ncol] = (__bf16)val;
      const float vv = vp[tok * Hh + ncol];
      float4 l0 = ((float4*)(sL + b * 8))[0], l1 = ((float4*)(sL + b * 8))[1];
      f0.x = f0.x * l0.x + vv * k0v.x;  f0.y = f0.y * l0.y + vv * k0v.y;
      f0.z = f0.z * l0.z + vv * k0v.z;  f0.w = f0.w * l0.w + vv * k0v.w;
      f1.x = f1.x * l1.x + vv * k1v.x;  f1.y = f1.y * l1.y + vv * k1v.y;
      f1.z = f1.z * l1.z + vv * k1v.z;  f1.w = f1.w * l1.w + vv * k1v.w;
      ((float4*)Fp)[0] = f0; ((float4*)Fp)[1] = f1;
    }

    // -------- grid-wide barrier between time steps --------
    __syncthreads();
    if (tid == 0) {
      __threadfence();
      volatile unsigned* vb = bar;
      unsigned gen = vb[1];
      unsigned arrived = atomicAdd(bar, 1u) + 1u;
      if (arrived == gridDim.x) {
        vb[0] = 0;
        __threadfence();
        atomicAdd(bar + 1, 1u);
      } else {
        while (vb[1] == gen) __builtin_amdgcn_s_sleep(2);
      }
      __threadfence();
    }
    __syncthreads();
  }
}

// -------------------------------------------------------------------- launcher
extern "C" void kernel_launch(void* const* d_in, const int* in_sizes, int n_in,
                              void* d_out, int out_size, void* d_ws, size_t ws_size,
                              hipStream_t stream) {
  (void)in_sizes; (void)n_in; (void)out_size; (void)ws_size;
  const float* x     = (const float*)d_in[0];
  const float* W_x   = (const float*)d_in[1];
  const float* W_h   = (const float*)d_in[2];
  const float* W_k   = (const float*)d_in[3];
  const float* W_v   = (const float*)d_in[4];
  const float* W_lam = (const float*)d_in[5];
  const float* bvec  = (const float*)d_in[6];
  const float* b_lam = (const float*)d_in[7];
  float* out = (float*)d_out;

  char* ws = (char*)d_ws;
  size_t off = 0;
  auto take = [&](size_t bytes) {
    char* p = ws + off;
    off = (off + bytes + 255) & ~(size_t)255;
    return p;
  };
  __bf16* xbf = (__bf16*)take((size_t)Mm * Ii * 2);
  __bf16* wxt = (__bf16*)take((size_t)Hh * Ii * 2);
  __bf16* wvt = (__bf16*)take((size_t)Hh * Ii * 2);
  __bf16* wht = (__bf16*)take((size_t)Hh * Hh * 2);
  float*  xp  = (float*)take((size_t)Mm * Hh * 4);
  float*  vp  = (float*)take((size_t)Mm * Hh * 4);
  float*  kraw= (float*)take((size_t)Mm * Rr * 4);
  float*  lraw= (float*)take((size_t)Mm * Rr * 4);
  float*  kn  = (float*)take((size_t)Mm * Rr * 4);
  float*  ln  = (float*)take((size_t)Mm * Rr * 4);
  __bf16* h0  = (__bf16*)take((size_t)Bb * Hh * 2);
  __bf16* h1  = (__bf16*)take((size_t)Bb * Hh * 2);
  float*  Fst = (float*)take((size_t)Bb * Hh * Rr * 4);
  unsigned* bar = (unsigned*)take(256);

  hipMemsetAsync(h0, 0, (size_t)Bb * Hh * 2, stream);
  hipMemsetAsync(h1, 0, (size_t)Bb * Hh * 2, stream);
  hipMemsetAsync(Fst, 0, (size_t)Bb * Hh * Rr * 4, stream);
  hipMemsetAsync(bar, 0, 256, stream);

  f32_to_bf16<<<4096, 256, 0, stream>>>(x, xbf, (size_t)Mm * Ii / 4);
  transpose_bf16<<<dim3(Ii / 32, Hh / 32), 256, 0, stream>>>(W_x, wxt, Ii, Hh);
  transpose_bf16<<<dim3(Ii / 32, Hh / 32), 256, 0, stream>>>(W_v, wvt, Ii, Hh);
  transpose_bf16<<<dim3(Hh / 32, Hh / 32), 256, 0, stream>>>(W_h, wht, Hh, Hh);

  gemm_bf16<<<dim3(Hh / 32, Mm / 64), 256, 0, stream>>>(xbf, wxt, bvec, xp, Hh, Ii);
  gemm_bf16<<<dim3(Hh / 32, Mm / 64), 256, 0, stream>>>(xbf, wvt, nullptr, vp, Hh, Ii);

  proj_small<<<Mm / 16, 256, 0, stream>>>(x, W_k, W_lam, kraw, lraw);
  finalize_klam<<<Mm / 256, 256, 0, stream>>>(kraw, lraw, b_lam, kn, ln);

  constexpr size_t SMEM = 32 * 2048 * 2 + 64 * 512 * 2 + 2 * 64 * 8 * 4; // 200704 B
  dream_rec<<<dim3(Hh / 32), 256, SMEM, stream>>>(wht, xp, vp, kn, ln, h0, h1, Fst,
                                                  out, bar);
}
